// seq_lstm_84181359001700
// MI455X (gfx1250) — matrix operand order repaired and checked by reading, not compiled, not run on hardware
//
#include <hip/hip_runtime.h>
#include <hip/hip_bf16.h>
#include <math.h>

// ---------------------------------------------------------------------------
// Time-aware LSTM scan, MI455X (gfx1250) wave32 + WMMA bf16.
//   B=2048, T=512, D=24, H=100, O=2
// One workgroup (8 wave32) owns a 16-row batch tile and runs the whole T=512
// recurrence. Weights live in VGPRs as pre-packed WMMA B-fragments; h/c state
// lives in LDS with bf16 shadow copies laid out for direct 16B A-frag loads.
// All A-frags for a GEMM are loaded before the WMMA chain so DS latency
// overlaps WMMA issue. Activations use gfx1250 hardware v_tanh_f32.
// 2 barriers per timestep.
// ---------------------------------------------------------------------------

typedef __attribute__((ext_vector_type(16))) __bf16 v16bf;
typedef __attribute__((ext_vector_type(8)))  __bf16 v8bf;
typedef __attribute__((ext_vector_type(8)))  float  v8f;

#define BATCH   2048
#define TSTEPS  512
#define DIN     24
#define HDIM    100
#define ODIM    2
#define MB      16            // batch rows per workgroup (WMMA M)
#define NWAVES  8
#define NTHREADS (NWAVES * 32)
#define HP      128           // H padded to 4 K-tiles of 32
#define GP      512           // 4 gates x 128 padded columns

#if __has_builtin(__builtin_amdgcn_tanhf)
__device__ __forceinline__ float hw_tanh(float x) { return __builtin_amdgcn_tanhf(x); }
#else
__device__ __forceinline__ float hw_tanh(float x) { return tanhf(x); }
#endif

// sigmoid(x) = 0.5*tanh(0.5x) + 0.5 : single TRANS op on gfx1250
__device__ __forceinline__ float fast_sigmoid(float x) {
  return __builtin_fmaf(0.5f, hw_tanh(0.5f * x), 0.5f);
}

// Build a 16x32 bf16 A-fragment from a bf16 LDS row (two contiguous 16B loads).
// Layout: lane row = lane&15; lanes 0-15: K = base+{0..7,16..23};
// lanes 16-31: K = base+{8..15,24..31}.  (khA = (lane>>4)*8)
__device__ __forceinline__ v16bf load_afrag(const __bf16* rowp, int base, int khA) {
  union { v16bf v; struct { v8bf lo, hi; } p; } u;
  u.p.lo = *(const v8bf*)(rowp + base + khA);        // 16B aligned
  u.p.hi = *(const v8bf*)(rowp + base + 16 + khA);   // 16B aligned
  return u.v;
}

__global__ void __launch_bounds__(NTHREADS, 1)
time_lstm_wmma_kernel(const float* __restrict__ x,      // [B,T,D]
                      const float* __restrict__ ts,     // [B,T]
                      const float* __restrict__ W_all,  // [H,4H] row-major
                      const float* __restrict__ b_all,  // [4H]
                      const float* __restrict__ U_all,  // [D,4H]
                      const float* __restrict__ b_u,    // [4H]
                      const float* __restrict__ W_d,    // [H,H]
                      const float* __restrict__ b_d,    // [H]
                      const float* __restrict__ W_lin,  // [H,O]
                      const float* __restrict__ b_lin,  // [O]
                      float* __restrict__ out)          // [B,O]
{
  __shared__ float  hlds[MB][HP];    // f32 hidden (for final head)
  __shared__ float  clds[MB][HP];    // f32 cell (master recurrence state)
  __shared__ __bf16 hbf[MB][HP];     // bf16 shadow of h (A-frag source)
  __shared__ __bf16 cbf[MB][HP];     // bf16 shadow of c (A-frag source)
  __shared__ float  cadj[MB][HP];    // time-adjusted cell
  __shared__ float  glds[MB][GP];    // sigmoided gates, padded layout
  __shared__ float  tslds[MB];       // per-row time value for current step

  const int tid  = threadIdx.x;
  const int lane = tid & 31;
  const int w    = tid >> 5;        // wave id 0..7
  const int lrow = lane & 15;       // A: row M ; B/C: col N within tile
  const int hif  = lane >> 4;       // lane half
  const int b0   = blockIdx.x * MB;

  // ---- zero all state (incl. padding columns) ----
  for (int i = tid; i < MB * HP; i += NTHREADS) {
    (&hlds[0][0])[i] = 0.0f;
    (&clds[0][0])[i] = 0.0f;
    (&hbf[0][0])[i]  = (__bf16)0.0f;
    (&cbf[0][0])[i]  = (__bf16)0.0f;
  }

  // =========================================================================
  // Pre-pack persistent bf16 B-fragments into registers (once).
  // B-matrix 32x16 layout (wave32): lanes 0-15 -> col N=lane, K={0..7,16..23};
  // lanes 16-31 -> col N=lane-16, K={8..15,24..31} (matches A-frag order).
  // =========================================================================
  const int khB = hif * 8;

  v16bf wallf[4][4];   // [nt][kt] : W_all fragments for this wave's 4 N-tiles
  v16bf uf[4];         // [nt]     : U_all fragments (single K tile, D=24<32)
  v16bf wdf[4];        // [kt]     : W_d fragments (this wave's single N-tile)
  float gbias[4];      // combined b_all + b_u for this lane's gate column
  float bdl;           // b_d for this lane's W_d column

  #pragma unroll
  for (int nt = 0; nt < 4; ++nt) {
    const int colp = w * 64 + nt * 16 + lrow;     // padded col in [0,512)
    const int g    = colp >> 7;                   // gate index 0..3
    const int j    = colp & 127;                  // col within gate
    const int rn   = (j < HDIM) ? (g * HDIM + j) : -1;   // real column in 4H
    gbias[nt] = (rn >= 0) ? (b_all[rn] + b_u[rn]) : 0.0f;
    #pragma unroll
    for (int kt = 0; kt < 4; ++kt) {
      #pragma unroll
      for (int e = 0; e < 16; ++e) {
        const int K = kt * 32 + khB + (e & 7) + ((e >> 3) << 4);
        const float v = (rn >= 0 && K < HDIM) ? W_all[K * (4 * HDIM) + rn] : 0.0f;
        wallf[nt][kt][e] = (__bf16)v;
      }
    }
    #pragma unroll
    for (int e = 0; e < 16; ++e) {
      const int K = khB + (e & 7) + ((e >> 3) << 4);
      const float v = (rn >= 0 && K < DIN) ? U_all[K * (4 * HDIM) + rn] : 0.0f;
      uf[nt][e] = (__bf16)v;
    }
  }
  {
    const int col = w * 16 + lrow;                // W_d col in [0,128)
    bdl = (col < HDIM) ? b_d[col] : 0.0f;
    #pragma unroll
    for (int kt = 0; kt < 4; ++kt) {
      #pragma unroll
      for (int e = 0; e < 16; ++e) {
        const int K = kt * 32 + khB + (e & 7) + ((e >> 3) << 4);
        const float v = (col < HDIM && K < HDIM) ? W_d[K * HDIM + col] : 0.0f;
        wdf[kt][e] = (__bf16)v;
      }
    }
  }

  // ts for step 0
  if (tid < MB) tslds[tid] = ts[(long long)(b0 + tid) * TSTEPS + 0];
  __syncthreads();

  const int khA = hif * 8;

  // =========================================================================
  // Recurrence over T
  // =========================================================================
  for (int t = 0; t < TSTEPS; ++t) {
    // ---- phase 1: c_s1 = tanh(c @ W_d + b_d); c_adj = (c-c_s1) + c_s1*t ----
    {
      v16bf ca[4];
      #pragma unroll
      for (int kt = 0; kt < 4; ++kt)      // issue all 8 ds_load_b128 first
        ca[kt] = load_afrag(&cbf[lrow][0], kt * 32, khA);
      v8f acc = {};
      #pragma unroll
      for (int kt = 0; kt < 4; ++kt)
        acc = __builtin_amdgcn_wmma_f32_16x16x32_bf16(
            false, ca[kt], false, wdf[kt], (short)0, acc, false, false);
      const int col = w * 16 + lrow;
      #pragma unroll
      for (int r = 0; r < 8; ++r) {
        const int row = r + 8 * hif;
        const float s1 = hw_tanh(acc[r] + bdl);
        const float cv = clds[row][col];
        cadj[row][col] = (cv - s1) + s1 * tslds[row];  // padding cols: 0
      }
    }

    // ---- phase 2: gates = sigmoid(h @ W_all + x_t @ U_all + bias) ----
    {
      v8f acc[4] = {};
      // x_t fragment: K = input feature (D=24, padded to 32)
      {
        v16bf a;
        const long long rbase = ((long long)(b0 + lrow) * TSTEPS + t) * DIN;
        if (t + 1 < TSTEPS) __builtin_prefetch(&x[rbase + DIN], 0, 1);
        #pragma unroll
        for (int v = 0; v < 8; ++v) {
          const int grp = v >> 2;
          const int k = grp * 16 + khA + (v & 3) * 2;
          const float v0 = (k     < DIN) ? x[rbase + k]     : 0.0f;
          const float v1 = (k + 1 < DIN) ? x[rbase + k + 1] : 0.0f;
          a[2 * v]     = (__bf16)v0;
          a[2 * v + 1] = (__bf16)v1;
        }
        #pragma unroll
        for (int nt = 0; nt < 4; ++nt)
          acc[nt] = __builtin_amdgcn_wmma_f32_16x16x32_bf16(
              false, a, false, uf[nt], (short)0, acc[nt], false, false);
      }
      // h @ W_all over K = 0..127 : load all frags, then WMMA chain
      {
        v16bf ha[4];
        #pragma unroll
        for (int kt = 0; kt < 4; ++kt)
          ha[kt] = load_afrag(&hbf[lrow][0], kt * 32, khA);
        #pragma unroll
        for (int kt = 0; kt < 4; ++kt)
          #pragma unroll
          for (int nt = 0; nt < 4; ++nt)
            acc[nt] = __builtin_amdgcn_wmma_f32_16x16x32_bf16(
                false, ha[kt], false, wallf[nt][kt], (short)0, acc[nt], false, false);
      }
      #pragma unroll
      for (int nt = 0; nt < 4; ++nt) {
        const int colp = w * 64 + nt * 16 + lrow;
        #pragma unroll
        for (int r = 0; r < 8; ++r) {
          const int row = r + 8 * hif;
          glds[row][colp] = fast_sigmoid(acc[nt][r] + gbias[nt]);
        }
      }
    }
    __syncthreads();   // gates + c_adj visible to all waves

    // ---- elementwise state update (f,i,o,c~ live in padded gate slots) ----
    for (int idx = tid; idx < MB * HDIM; idx += NTHREADS) {
      const int row = idx / HDIM;
      const int j   = idx - row * HDIM;
      const float fg = glds[row][j];
      const float ig = glds[row][128 + j];
      const float og = glds[row][256 + j];
      const float ct = glds[row][384 + j];
      const float cn = fg * cadj[row][j] + ig * ct;
      const float hn = og * hw_tanh(cn);
      clds[row][j] = cn;
      hlds[row][j] = hn;
      cbf[row][j]  = (__bf16)cn;   // bf16 shadows for next step's A-frags
      hbf[row][j]  = (__bf16)hn;
    }
    // stage ts for next step (current tslds reads completed before barrier C)
    if (tid < MB && t + 1 < TSTEPS)
      tslds[tid] = ts[(long long)(b0 + tid) * TSTEPS + (t + 1)];
    __syncthreads();   // h, c, shadows, ts visible for next iteration
  }

  // ---- output head: relu(h_last @ W_lin + b_lin), O=2 ----
  if (tid < MB * ODIM) {
    const int row = tid >> 1;
    const int o   = tid & 1;
    float s = b_lin[o];
    #pragma unroll 4
    for (int k = 0; k < HDIM; ++k) s += hlds[row][k] * W_lin[k * ODIM + o];
    out[(long long)(b0 + row) * ODIM + o] = fmaxf(s, 0.0f);
  }
}

extern "C" void kernel_launch(void* const* d_in, const int* in_sizes, int n_in,
                              void* d_out, int out_size, void* d_ws, size_t ws_size,
                              hipStream_t stream) {
  (void)in_sizes; (void)n_in; (void)d_ws; (void)ws_size; (void)out_size;
  const float* x      = (const float*)d_in[0];
  const float* ts     = (const float*)d_in[1];
  const float* W_all  = (const float*)d_in[2];
  const float* b_all  = (const float*)d_in[3];
  const float* U_all  = (const float*)d_in[4];
  const float* b_u    = (const float*)d_in[5];
  const float* W_d    = (const float*)d_in[6];
  const float* b_d    = (const float*)d_in[7];
  const float* W_lin  = (const float*)d_in[8];
  const float* b_lin  = (const float*)d_in[9];
  float* out = (float*)d_out;

  dim3 grid(BATCH / MB);        // 128 workgroups
  dim3 block(NTHREADS);         // 256 threads = 8 wave32
  time_lstm_wmma_kernel<<<grid, block, 0, stream>>>(
      x, ts, W_all, b_all, U_all, b_u, W_d, b_d, W_lin, b_lin, out);
}
